// AttentionHead_2250562863203
// MI455X (gfx1250) — compile-verified, matrix operand order checked by reading, not run on hardware
//
#include <hip/hip_runtime.h>
#include <hip/hip_fp16.h>
#include <math.h>

#define DMODEL 1024
#define SEQ    2048
#define BATCH  4
#define ROWS   (BATCH * SEQ) // 8192

typedef __attribute__((ext_vector_type(16))) _Float16 v16h;
typedef __attribute__((ext_vector_type(8)))  _Float16 v8h;
typedef __attribute__((ext_vector_type(4)))  _Float16 v4h;
typedef __attribute__((ext_vector_type(8)))  float    v8f;

// ---------------------------------------------------------------------------
// CDNA5 WMMA: D = A(16x32 f16) * B(32x16 f16) + C(16x16 f32), wave32.
// ---------------------------------------------------------------------------
static __device__ __forceinline__ v8f wmma16x16x32(v16h a, v16h b, v8f c) {
  return __builtin_amdgcn_wmma_f32_16x16x32_f16(
      /*neg_a=*/false, a, /*neg_b=*/false, b,
      /*c_mod=*/(short)0, c, /*reuse_a=*/false, /*reuse_b=*/false);
}

// Load a per-lane A/B fragment chunk for 16-bit 16x32 (or 32x16) operands.
// Per ISA 7.12.2: lanes 0-15 hold K = [0..7] (v0-3) and [16..23] (v4-7);
// lanes 16-31 hold K = [8..15] and [24..31].  Caller passes p already offset
// by row*ldk + k0 + koff, with koff = (lane>=16 ? 8 : 0).  Both 8-half chunks
// are 16B-aligned -> global_load_b128 x2.
static __device__ __forceinline__ v16h load_frag(const _Float16* __restrict__ p) {
  v8h lo = *(const v8h*)(p);
  v8h hi = *(const v8h*)(p + 16);
  v16h r;
#pragma unroll
  for (int i = 0; i < 8; ++i) { r[i] = lo[i]; r[i + 8] = hi[i]; }
  return r;
}

// ---------------------------------------------------------------------------
// Kernel 1: f32 -> f16 conversion of x (8M elements, 4 per thread).
// ---------------------------------------------------------------------------
__global__ void cvt_x_kernel(const float* __restrict__ in,
                             _Float16* __restrict__ out, int n) {
  int i = (blockIdx.x * blockDim.x + threadIdx.x) * 4;
  if (i < n) {
    float4 f = *(const float4*)(in + i);
    v4h h;
    h[0] = (_Float16)f.x; h[1] = (_Float16)f.y;
    h[2] = (_Float16)f.z; h[3] = (_Float16)f.w;
    *(v4h*)(out + i) = h;
  }
}

// ---------------------------------------------------------------------------
// Kernel 2: W[K][N] f32 -> Wt[N][K] f16 (transpose + convert), 32x32 tiles.
// ---------------------------------------------------------------------------
__global__ void transpose_w_kernel(const float* __restrict__ W,
                                   _Float16* __restrict__ Wt) {
  __shared__ float tile[32][33];
  int n0 = blockIdx.x * 32, k0 = blockIdx.y * 32;
  int tx = threadIdx.x, ty = threadIdx.y; // block (32, 8)
#pragma unroll
  for (int i = 0; i < 32; i += 8)
    tile[ty + i][tx] = W[(size_t)(k0 + ty + i) * DMODEL + n0 + tx];
  __syncthreads();
#pragma unroll
  for (int i = 0; i < 32; i += 8)
    Wt[(size_t)(n0 + ty + i) * DMODEL + k0 + tx] = (_Float16)tile[tx][ty + i];
}

// ---------------------------------------------------------------------------
// Kernel 3: QKV projection GEMM.  Out[m][n] = sum_k Xh[m][k]*Wt[n][k] + b[n].
// Block = 256 threads = 8 waves; block tile 128(M) x 128(N); wave tile 32x64.
// TRANSPOSED_OUT=true stores per-batch transposed ([b][n][s]) for the V path.
// ---------------------------------------------------------------------------
template <bool TRANSPOSED_OUT>
__global__ void qkv_gemm_kernel(const _Float16* __restrict__ Xh,  // [8192][1024]
                                const _Float16* __restrict__ Wt,  // [1024][1024]
                                const float* __restrict__ bias,   // [1024]
                                _Float16* __restrict__ Out) {
  const int lane = threadIdx.x & 31;
  const int wave = threadIdx.x >> 5;
  const int l15  = lane & 15;
  const int koff = (lane >> 4) << 3; // 0 or 8
  const int Mbase = blockIdx.x * 128 + (wave & 3) * 32;
  const int Nbase = blockIdx.y * 128 + (wave >> 2) * 64;

  v8f acc[2][4];
#pragma unroll
  for (int tm = 0; tm < 2; ++tm)
#pragma unroll
    for (int tn = 0; tn < 4; ++tn) acc[tm][tn] = (v8f)0.0f;

  const _Float16* aBase = Xh + (size_t)(Mbase + l15) * DMODEL + koff;
  const _Float16* bBase = Wt + (size_t)(Nbase + l15) * DMODEL + koff;

  for (int k0 = 0; k0 < DMODEL; k0 += 32) {
    v16h a[2], b[4];
#pragma unroll
    for (int tm = 0; tm < 2; ++tm)
      a[tm] = load_frag(aBase + (size_t)tm * 16 * DMODEL + k0);
#pragma unroll
    for (int tn = 0; tn < 4; ++tn)
      b[tn] = load_frag(bBase + (size_t)tn * 16 * DMODEL + k0);
#pragma unroll
    for (int tm = 0; tm < 2; ++tm)
#pragma unroll
      for (int tn = 0; tn < 4; ++tn)
        acc[tm][tn] = wmma16x16x32(a[tm], b[tn], acc[tm][tn]);
  }

  // Epilogue: bias add, convert to f16, store.
  const int rbase = (lane >> 4) << 3; // C layout: row = vgpr + (lane>=16 ? 8:0)
#pragma unroll
  for (int tm = 0; tm < 2; ++tm) {
#pragma unroll
    for (int tn = 0; tn < 4; ++tn) {
      const int col = Nbase + tn * 16 + l15;
      const float bv = bias[col];
#pragma unroll
      for (int r = 0; r < 8; ++r) {
        const int m = Mbase + tm * 16 + rbase + r;
        const float v = acc[tm][tn][r] + bv;
        if (!TRANSPOSED_OUT) {
          Out[(size_t)m * DMODEL + col] = (_Float16)v;
        } else {
          const int bb = m >> 11;        // batch
          const int s  = m & (SEQ - 1);  // position
          Out[((size_t)(bb * DMODEL + col) << 11) + s] = (_Float16)v;
        }
      }
    }
  }
}

// ---------------------------------------------------------------------------
// Kernel 4: fused attention for one (batch, 16-query tile).
// Phase 1: S[16][2048] = Q*K^T / 32 into LDS (8 waves x 256 keys each).
// Phase 2: row softmax in f32 (in place).
// Phase 3: O[16][1024] = P * V, waves split over d (128 cols each).
// ---------------------------------------------------------------------------
__global__ void attn_kernel(const _Float16* __restrict__ Qh,  // [b*s][1024]
                            const _Float16* __restrict__ Kh,  // [b*s][1024]
                            const _Float16* __restrict__ Vt,  // [b][1024][2048]
                            float* __restrict__ Out) {        // [b][s][1024]
  __shared__ float sS[16 * SEQ];        // 128 KiB score tile
  __shared__ float sRedMax[16 * 16];
  __shared__ float sRedSum[16 * 16];

  const int b     = blockIdx.y;
  const int qbase = blockIdx.x * 16;
  const int lane  = threadIdx.x & 31;
  const int wave  = threadIdx.x >> 5;
  const int l15   = lane & 15;
  const int koff  = (lane >> 4) << 3;
  const int rbase = (lane >> 4) << 3;

  const _Float16* Qb = Qh + (size_t)b * SEQ * DMODEL;
  const _Float16* Kb = Kh + (size_t)b * SEQ * DMODEL;
  const _Float16* Vb = Vt + (size_t)b * DMODEL * SEQ;

  // ---- Phase 1: scores ----------------------------------------------------
  const _Float16* qRow = Qb + (size_t)(qbase + l15) * DMODEL + koff;
  const int keyWaveBase = wave * 256;
#pragma unroll
  for (int ktc = 0; ktc < 4; ++ktc) { // 4 chunks x 4 key tiles = 256 keys
    v8f acc[4];
#pragma unroll
    for (int t = 0; t < 4; ++t) acc[t] = (v8f)0.0f;
    for (int k0 = 0; k0 < DMODEL; k0 += 32) {
      v16h a = load_frag(qRow + k0); // shared across the 4 key tiles
#pragma unroll
      for (int t = 0; t < 4; ++t) {
        const int keyBase = keyWaveBase + (ktc * 4 + t) * 16;
        v16h bf = load_frag(Kb + (size_t)(keyBase + l15) * DMODEL + k0 + koff);
        acc[t] = wmma16x16x32(a, bf, acc[t]);
      }
    }
#pragma unroll
    for (int t = 0; t < 4; ++t) {
      const int col = keyWaveBase + (ktc * 4 + t) * 16 + l15;
#pragma unroll
      for (int r = 0; r < 8; ++r)
        sS[(rbase + r) * SEQ + col] = acc[t][r] * 0.03125f; // 1/sqrt(1024)
    }
  }
  __syncthreads();

  // ---- Phase 2: softmax (256 threads: 16 rows x 16 segments of 128) -------
  {
    const int r   = threadIdx.x & 15;
    const int seg = threadIdx.x >> 4;
    float* row = &sS[r * SEQ + seg * 128];
    float mx = -INFINITY;
    for (int i = 0; i < 128; ++i) mx = fmaxf(mx, row[i]);
    sRedMax[r * 16 + seg] = mx;
    __syncthreads();
    mx = -INFINITY;
#pragma unroll
    for (int i = 0; i < 16; ++i) mx = fmaxf(mx, sRedMax[r * 16 + i]);
    float sum = 0.0f;
    for (int i = 0; i < 128; ++i) {
      const float e = __expf(row[i] - mx);
      row[i] = e;
      sum += e;
    }
    sRedSum[r * 16 + seg] = sum;
    __syncthreads();
    sum = 0.0f;
#pragma unroll
    for (int i = 0; i < 16; ++i) sum += sRedSum[r * 16 + i];
    const float inv = 1.0f / sum;
    for (int i = 0; i < 128; ++i) row[i] *= inv;
  }
  __syncthreads();

  // ---- Phase 3: O = P * V, wave handles d-chunk of 128 --------------------
  const int dbase = wave * 128;
  v8f acc[8];
#pragma unroll
  for (int t = 0; t < 8; ++t) acc[t] = (v8f)0.0f;

  for (int kc0 = 0; kc0 < SEQ; kc0 += 32) {
    // Rebuild f16 A fragment from f32 LDS probabilities (ds_load_b128 x4).
    const float* s0 = &sS[l15 * SEQ + kc0 + koff];
    const float4 f0 = *(const float4*)(s0);
    const float4 f1 = *(const float4*)(s0 + 4);
    const float4 f2 = *(const float4*)(s0 + 16);
    const float4 f3 = *(const float4*)(s0 + 20);
    v16h a;
    a[0]  = (_Float16)f0.x; a[1]  = (_Float16)f0.y; a[2]  = (_Float16)f0.z; a[3]  = (_Float16)f0.w;
    a[4]  = (_Float16)f1.x; a[5]  = (_Float16)f1.y; a[6]  = (_Float16)f1.z; a[7]  = (_Float16)f1.w;
    a[8]  = (_Float16)f2.x; a[9]  = (_Float16)f2.y; a[10] = (_Float16)f2.z; a[11] = (_Float16)f2.w;
    a[12] = (_Float16)f3.x; a[13] = (_Float16)f3.y; a[14] = (_Float16)f3.z; a[15] = (_Float16)f3.w;
#pragma unroll
    for (int tn = 0; tn < 8; ++tn) {
      const _Float16* p =
          Vb + (size_t)(dbase + tn * 16 + l15) * SEQ + kc0 + koff;
      v16h bf = load_frag(p);
      acc[tn] = wmma16x16x32(a, bf, acc[tn]);
    }
  }

  float* Ob = Out + ((size_t)b * SEQ + qbase) * DMODEL;
#pragma unroll
  for (int tn = 0; tn < 8; ++tn) {
    const int col = dbase + tn * 16 + l15;
#pragma unroll
    for (int r = 0; r < 8; ++r)
      Ob[(size_t)(rbase + r) * DMODEL + col] = acc[tn][r];
  }
}

// ---------------------------------------------------------------------------
// Host-side launcher.
// ---------------------------------------------------------------------------
extern "C" void kernel_launch(void* const* d_in, const int* in_sizes, int n_in,
                              void* d_out, int out_size, void* d_ws, size_t ws_size,
                              hipStream_t stream) {
  const float* x  = (const float*)d_in[0];
  const float* Wq = (const float*)d_in[1];
  const float* bq = (const float*)d_in[2];
  const float* Wk = (const float*)d_in[3];
  const float* bk = (const float*)d_in[4];
  const float* Wv = (const float*)d_in[5];
  const float* bv = (const float*)d_in[6];
  float* out = (float*)d_out;

  char* ws = (char*)d_ws;
  _Float16* xh  = (_Float16*)(ws);                           // 16 MiB
  _Float16* wqt = (_Float16*)(ws + ((size_t)16 << 20));      //  2 MiB
  _Float16* wkt = (_Float16*)(ws + ((size_t)18 << 20));      //  2 MiB
  _Float16* wvt = (_Float16*)(ws + ((size_t)20 << 20));      //  2 MiB
  _Float16* qh  = (_Float16*)(ws + ((size_t)22 << 20));      // 16 MiB
  _Float16* kh  = (_Float16*)(ws + ((size_t)38 << 20));      // 16 MiB
  _Float16* vt  = (_Float16*)(ws + ((size_t)54 << 20));      // 16 MiB

  // 1) Convert x to f16.
  {
    const int n = ROWS * DMODEL; // 8388608
    cvt_x_kernel<<<n / (256 * 4), 256, 0, stream>>>(x, xh, n);
  }
  // 2) Transpose + convert weights to [N][K] f16.
  {
    dim3 grid(DMODEL / 32, DMODEL / 32), block(32, 8);
    transpose_w_kernel<<<grid, block, 0, stream>>>(Wq, wqt);
    transpose_w_kernel<<<grid, block, 0, stream>>>(Wk, wkt);
    transpose_w_kernel<<<grid, block, 0, stream>>>(Wv, wvt);
  }
  // 3) QKV projections (V stored transposed per batch).
  {
    dim3 grid(ROWS / 128, DMODEL / 128); // 64 x 8
    qkv_gemm_kernel<false><<<grid, 256, 0, stream>>>(xh, wqt, bq, qh);
    qkv_gemm_kernel<false><<<grid, 256, 0, stream>>>(xh, wkt, bk, kh);
    qkv_gemm_kernel<true ><<<grid, 256, 0, stream>>>(xh, wvt, bv, vt);
  }
  // 4) Fused attention.
  {
    dim3 grid(SEQ / 16, BATCH); // 128 x 4
    attn_kernel<<<grid, 256, 0, stream>>>(qh, kh, vt, out);
  }
  (void)in_sizes; (void)n_in; (void)out_size; (void)ws_size;
}
